// SelfAttention_14671608283741
// MI455X (gfx1250) — compile-verified, matrix operand order checked by reading, not run on hardware
//
#include <hip/hip_runtime.h>
#include <hip/hip_bf16.h>

// ---------------------------------------------------------------------------
// Self-attention forward for MI455X (gfx1250), bf16 WMMA path.
// B=8, T=1024, K=1024, H=16, s=64.
// ---------------------------------------------------------------------------

typedef __bf16 bf16;
typedef __attribute__((ext_vector_type(16))) __bf16 v16bf;
typedef __attribute__((ext_vector_type(4)))  __bf16 v4bf;
typedef __attribute__((ext_vector_type(8)))  float  v8f;

#define B_DIM 8
#define T_DIM 1024
#define K_DIM 1024
#define H_DIM 16
#define S_DIM 64

__device__ inline v8f wmma_bf16(v16bf a, v16bf b, v8f c) {
  // D = A(16x32) * B(32x16) + C, f32 accumulate
  return __builtin_amdgcn_wmma_f32_16x16x32_bf16(
      /*neg_a=*/false, a, /*neg_b=*/false, b,
      /*c_mod=*/(short)0, c, /*reuse_a=*/false, /*reuse_b=*/false);
}

// A-layout 16x32 bf16 fragment from a row-major matrix (leading dim ld).
// Per ISA 7.12.2: lanes 0-15 hold M=lane with K={0..7,16..23};
// lanes 16-31 hold M=lane-16 with K={8..15,24..31}.
// Each lane's data = two contiguous 16-byte chunks -> two b128 loads.
// Also serves as the B fragment when B[k][n] = src[n*ld + k] (B = src^T),
// since the B lane->(n,k) mapping mirrors the A lane->(m,k) mapping.
__device__ inline v16bf load_frag_row(const bf16* base, int ld, int row0, int k0) {
  const int lane = threadIdx.x & 31;
  const bf16* p = base + (size_t)(row0 + (lane & 15)) * ld + k0 + ((lane >> 4) << 3);
  v16bf r;
  reinterpret_cast<float4*>(&r)[0] = *reinterpret_cast<const float4*>(p);
  reinterpret_cast<float4*>(&r)[1] = *reinterpret_cast<const float4*>(p + 16);
  return r;
}

// ---------------------------------------------------------------------------
// Kernel 1: f32 -> bf16 conversion (grid-stride)
// ---------------------------------------------------------------------------
__global__ void cvt_f32_bf16(const float* __restrict__ src, bf16* __restrict__ dst, int n) {
  for (int i = blockIdx.x * blockDim.x + threadIdx.x; i < n; i += gridDim.x * blockDim.x)
    dst[i] = (bf16)src[i];
}

// ---------------------------------------------------------------------------
// Kernel 2: fused QKV projection.  out[b,t,j] = sum_k x[b,t,k] * W[j,k]
// (torch Linear: x @ W^T). M = B*T = 8192, N = 1024, K-loop = 1024.
// Block: 256 threads = 8 waves, tile 256(M) x 64(N); wave -> 32 x 64
// (B fragments reused across two M-tiles: 8 WMMA per 12 b128 loads).
// Q,K written [b,h,t,s]; V written TRANSPOSED [b,h,s,t] so the attention
// P@V B-fragments become contiguous b128 loads. blockIdx.z selects Q/K/V.
// ---------------------------------------------------------------------------
__global__ __launch_bounds__(256) void qkv_proj(
    const bf16* __restrict__ xb,
    const bf16* __restrict__ Wq, const bf16* __restrict__ Wk, const bf16* __restrict__ Wv,
    bf16* __restrict__ qo, bf16* __restrict__ ko, bf16* __restrict__ vo) {
  const bf16* W;
  bf16* out;
  if (blockIdx.z == 0)      { W = Wq; out = qo; }
  else if (blockIdx.z == 1) { W = Wk; out = ko; }
  else                      { W = Wv; out = vo; }
  const bool vt = (blockIdx.z == 2);

  const int wave = threadIdx.x >> 5;
  const int lane = threadIdx.x & 31;
  const int row0 = blockIdx.x * 256 + wave * 32;  // M index (b*T + t)
  const int col0 = blockIdx.y * 64;               // N index (j)

  v8f acc[2][4] = {};
  for (int k0 = 0; k0 < K_DIM; k0 += 32) {
    if (k0 + 32 < K_DIM) {  // prefetch next A k-tile -> global_prefetch_b8
      __builtin_prefetch(xb + (size_t)(row0 + (lane & 15)) * K_DIM + k0 + 32, 0, 1);
      __builtin_prefetch(xb + (size_t)(row0 + 16 + (lane & 15)) * K_DIM + k0 + 32, 0, 1);
    }
    v16bf a0 = load_frag_row(xb, K_DIM, row0,      k0);
    v16bf a1 = load_frag_row(xb, K_DIM, row0 + 16, k0);
#pragma unroll
    for (int nt = 0; nt < 4; ++nt) {
      v16bf b = load_frag_row(W, K_DIM, col0 + nt * 16, k0);  // B = W^T
      acc[0][nt] = wmma_bf16(a0, b, acc[0][nt]);
      acc[1][nt] = wmma_bf16(a1, b, acc[1][nt]);
    }
  }

  const int rbase = (lane >> 4) << 3;  // lanes 16-31 hold rows M+8..M+15
  const int n = lane & 15;
#pragma unroll
  for (int mi = 0; mi < 2; ++mi) {
#pragma unroll
    for (int nt = 0; nt < 4; ++nt) {
#pragma unroll
      for (int r = 0; r < 8; ++r) {
        const int m = row0 + mi * 16 + rbase + r;
        const int b = m >> 10, t = m & 1023;
        const int j = col0 + nt * 16 + n;
        const int h = j >> 6, s = j & 63;
        const bf16 val = (bf16)acc[mi][nt][r];
        const size_t head = ((size_t)((b << 4) + h)) << 16;  // (b*16+h)*65536
        if (vt)
          out[head + ((size_t)s << 10) + t] = val;   // V: [b,h,s,t]
        else
          out[head + ((size_t)t << 6) + s] = val;    // Q,K: [b,h,t,s]
      }
    }
  }
}

// ---------------------------------------------------------------------------
// Kernel 3: attention for one (b,h), 64 query rows per block.
// Block: 128 threads = 4 waves (16 query rows each).
// Phase 1: S = Q K^T (16x1024 per wave) -> LDS f32 (64*1024*4 = 256 KB dyn).
// Phase 2: per-row max + unnormalized exp, float4-vectorized, compacted
//          in-place to bf16 (write byte 8i <= read byte 16i: safe);
//          1/sum folded into the epilogue.
// Phase 3: O = P V via WMMA; P A-frags from LDS (ds_load_b128), V B-frags
//          from the transposed [b,h,s,t] layout (global b128).
// ---------------------------------------------------------------------------
__global__ __launch_bounds__(128) void attn_kernel(
    const bf16* __restrict__ qb, const bf16* __restrict__ kb,
    const bf16* __restrict__ vtb, bf16* __restrict__ yb) {
  extern __shared__ float sS[];   // 64 * 1024 floats
  __shared__ float sInv[64];

  const int bh = blockIdx.y;              // b*16 + h
  const int b = bh >> 4, h = bh & 15;
  const bf16* Q  = qb  + ((size_t)bh << 16);   // 1024*64 per head
  const bf16* Km = kb  + ((size_t)bh << 16);
  const bf16* Vt = vtb + ((size_t)bh << 16);   // [s, t] row-major, ld = 1024

  const int wave = threadIdx.x >> 5;
  const int lane = threadIdx.x & 31;
  const int qrow0 = blockIdx.x * 64 + wave * 16;
  const int rbase = (lane >> 4) << 3;
  const int n = lane & 15;

  // Q fragments for this wave's 16 rows (K=64 -> two 16x32 A frags)
  v16bf a0 = load_frag_row(Q, S_DIM, qrow0, 0);
  v16bf a1 = load_frag_row(Q, S_DIM, qrow0, 32);

  // ---- Phase 1: scores into LDS ----
  for (int j0 = 0; j0 < T_DIM; j0 += 16) {
    // B = K^T: B[k][n] = K[j0+n][k] -> A-style load from K rows
    v16bf b0 = load_frag_row(Km, S_DIM, j0, 0);
    v16bf b1 = load_frag_row(Km, S_DIM, j0, 32);
    v8f c = {};
    c = wmma_bf16(a0, b0, c);
    c = wmma_bf16(a1, b1, c);
    const int rl = wave * 16 + rbase;
#pragma unroll
    for (int r = 0; r < 8; ++r)
      sS[(size_t)(rl + r) * 1024 + j0 + n] = c[r];
  }
  __syncthreads();

  // ---- Phase 2: softmax (vectorized; unnormalized exp -> bf16 in place) ----
  if (threadIdx.x < 64) {
    float* row = sS + (size_t)threadIdx.x * 1024;
    const float4* row4 = reinterpret_cast<const float4*>(row);
    float mx = -3.402823466e38f;
    for (int i = 0; i < 256; ++i) {
      const float4 v = row4[i];
      mx = fmaxf(mx, fmaxf(fmaxf(v.x, v.y), fmaxf(v.z, v.w)));
    }
    float sum = 0.f;
    v4bf* prow = reinterpret_cast<v4bf*>(row);  // 8-byte writes at 8i, reads at 16i
    for (int i = 0; i < 256; ++i) {
      const float4 v = row4[i];
      const float e0 = __expf(v.x - mx), e1 = __expf(v.y - mx);
      const float e2 = __expf(v.z - mx), e3 = __expf(v.w - mx);
      sum += (e0 + e1) + (e2 + e3);
      v4bf p;
      p[0] = (bf16)e0; p[1] = (bf16)e1; p[2] = (bf16)e2; p[3] = (bf16)e3;
      prow[i] = p;
    }
    sInv[threadIdx.x] = 1.f / sum;
  }
  __syncthreads();

  // ---- Phase 3: O = P @ V ----
  const bf16* P = reinterpret_cast<const bf16*>(sS);  // row stride 2048 bf16
  v8f o[4] = {};
  for (int k0 = 0; k0 < T_DIM; k0 += 32) {
    v16bf a = load_frag_row(P, 2048, wave * 16, k0);   // LDS -> ds_load_b128
#pragma unroll
    for (int nt = 0; nt < 4; ++nt) {
      // B[k][n] = V[k][n] = Vt[n*1024 + k] -> A-style load from Vt rows
      v16bf bb = load_frag_row(Vt, T_DIM, nt * 16, k0);
      o[nt] = wmma_bf16(a, bb, o[nt]);
    }
  }

#pragma unroll
  for (int nt = 0; nt < 4; ++nt) {
#pragma unroll
    for (int r = 0; r < 8; ++r) {
      const int rl = wave * 16 + rbase + r;
      const int t = blockIdx.x * 64 + rl;
      const float val = o[nt][r] * sInv[rl];
      // y[b,t,K] layout
      yb[(((size_t)b << 10) + t) * 1024 + (h << 6) + nt * 16 + n] = (bf16)val;
    }
  }
}

// ---------------------------------------------------------------------------
// Kernel 4: output projection. out[m,j] = sum_k y[m,k]*Wu[j,k] + bu[j] (f32)
// Same 32x64 wave tile as kernel 2.
// ---------------------------------------------------------------------------
__global__ __launch_bounds__(256) void out_proj(
    const bf16* __restrict__ yb, const bf16* __restrict__ Wu,
    const float* __restrict__ bu, float* __restrict__ out) {
  const int wave = threadIdx.x >> 5;
  const int lane = threadIdx.x & 31;
  const int row0 = blockIdx.x * 256 + wave * 32;
  const int col0 = blockIdx.y * 64;

  v8f acc[2][4] = {};
  for (int k0 = 0; k0 < K_DIM; k0 += 32) {
    if (k0 + 32 < K_DIM) {
      __builtin_prefetch(yb + (size_t)(row0 + (lane & 15)) * K_DIM + k0 + 32, 0, 1);
      __builtin_prefetch(yb + (size_t)(row0 + 16 + (lane & 15)) * K_DIM + k0 + 32, 0, 1);
    }
    v16bf a0 = load_frag_row(yb, K_DIM, row0,      k0);
    v16bf a1 = load_frag_row(yb, K_DIM, row0 + 16, k0);
#pragma unroll
    for (int nt = 0; nt < 4; ++nt) {
      v16bf b = load_frag_row(Wu, K_DIM, col0 + nt * 16, k0);  // B = Wu^T
      acc[0][nt] = wmma_bf16(a0, b, acc[0][nt]);
      acc[1][nt] = wmma_bf16(a1, b, acc[1][nt]);
    }
  }

  const int rbase = (lane >> 4) << 3;
  const int n = lane & 15;
#pragma unroll
  for (int mi = 0; mi < 2; ++mi) {
#pragma unroll
    for (int nt = 0; nt < 4; ++nt) {
#pragma unroll
      for (int r = 0; r < 8; ++r) {
        const int m = row0 + mi * 16 + rbase + r;
        const int j = col0 + nt * 16 + n;
        out[(size_t)m * 1024 + j] = acc[mi][nt][r] + bu[j];
      }
    }
  }
}

// ---------------------------------------------------------------------------
// Host launch
// ---------------------------------------------------------------------------
extern "C" void kernel_launch(void* const* d_in, const int* in_sizes, int n_in,
                              void* d_out, int out_size, void* d_ws, size_t ws_size,
                              hipStream_t stream) {
  const float* x  = (const float*)d_in[0];
  const float* Wq = (const float*)d_in[1];
  const float* Wk = (const float*)d_in[2];
  const float* Wv = (const float*)d_in[3];
  const float* Wu = (const float*)d_in[4];
  const float* bu = (const float*)d_in[5];
  float* out = (float*)d_out;

  char* ws = (char*)d_ws;
  const size_t XB   = (size_t)B_DIM * T_DIM * K_DIM * sizeof(bf16);  // 16 MB
  const size_t WB   = (size_t)K_DIM * K_DIM * sizeof(bf16);          // 2 MB
  const size_t HEAD = (size_t)B_DIM * H_DIM * T_DIM * S_DIM * sizeof(bf16); // 16 MB
  bf16* xb  = (bf16*)(ws);
  bf16* Wqb = (bf16*)(ws + XB);
  bf16* Wkb = (bf16*)(ws + XB + WB);
  bf16* Wvb = (bf16*)(ws + XB + 2 * WB);
  bf16* Wub = (bf16*)(ws + XB + 3 * WB);
  bf16* qbf = (bf16*)(ws + XB + 4 * WB);
  bf16* kbf = (bf16*)(ws + XB + 4 * WB + HEAD);
  bf16* vbf = (bf16*)(ws + XB + 4 * WB + 2 * HEAD);   // transposed [b,h,s,t]
  bf16* ybf = (bf16*)(ws + XB + 4 * WB + 3 * HEAD);
  // total workspace use: 16 + 8 + 48 + 16 = 88 MB = 92,274,688 bytes

  // 1) convert inputs to bf16
  const int nX = B_DIM * T_DIM * K_DIM;
  const int nW = K_DIM * K_DIM;
  cvt_f32_bf16<<<4096, 256, 0, stream>>>(x, xb, nX);
  cvt_f32_bf16<<<1024, 256, 0, stream>>>(Wq, Wqb, nW);
  cvt_f32_bf16<<<1024, 256, 0, stream>>>(Wk, Wkb, nW);
  cvt_f32_bf16<<<1024, 256, 0, stream>>>(Wv, Wvb, nW);
  cvt_f32_bf16<<<1024, 256, 0, stream>>>(Wu, Wub, nW);

  // 2) fused QKV projection (z selects q/k/v; v stored transposed)
  {
    dim3 grid(32, 16, 3);  // M/256, N/64
    qkv_proj<<<grid, 256, 0, stream>>>(xb, Wqb, Wkb, Wvb, qbf, kbf, vbf);
  }

  // 3) attention: 16 query blocks x 128 (b,h) pairs, 256 KB dynamic LDS
  {
    const size_t smem = (size_t)64 * 1024 * sizeof(float);  // 262144
    hipFuncSetAttribute((const void*)attn_kernel,
                        hipFuncAttributeMaxDynamicSharedMemorySize, (int)smem);
    dim3 grid(16, 128);
    attn_kernel<<<grid, 128, smem, stream>>>(qbf, kbf, vbf, ybf);
  }

  // 4) output projection + bias -> f32
  {
    dim3 grid(32, 16);
    out_proj<<<grid, 256, 0, stream>>>(ybf, Wub, bu, out);
  }
}
